// Attention_preprocessor_85272280695437
// MI455X (gfx1250) — matrix operand order repaired and checked by reading, not compiled, not run on hardware
//
#include <hip/hip_runtime.h>

typedef __attribute__((ext_vector_type(16))) _Float16 v16h;
typedef __attribute__((ext_vector_type(8)))  _Float16 v8h;
typedef __attribute__((ext_vector_type(2)))  _Float16 v2h;
typedef __attribute__((ext_vector_type(8)))  float    v8f;

constexpr int kNA   = 32;   // nodes per batch element
constexpr int kDIN  = 8;
constexpr int kH    = 64;   // hidden
constexpr int kDOUT = 8;
constexpr int kDinPad = 32; // x / W_in K-dim padded to one WMMA K step
constexpr int kOutPad = 16; // W_out M rows padded to one WMMA N tile

// f16 weight workspace layout (element offsets)
constexpr int kOffAq  = 0;
constexpr int kOffAk  = kH * kH;
constexpr int kOffAv  = 2 * kH * kH;
constexpr int kOffWin = 3 * kH * kH;                 // 64 x 32 (cols 8..31 zero)
constexpr int kOffWout= 3 * kH * kH + kH * kDinPad;  // 16 x 64 (rows 8..15 zero)

// silu(x) = x * sigmoid(x); v_rcp_f32 instead of IEEE division expansion.
__device__ __forceinline__ float silu_f(float v) {
    return v * __builtin_amdgcn_rcpf(1.0f + __expf(-v));
}

// ---------------------------------------------------------------------------
// Fragment helpers for v_wmma_f32_16x16x32_f16 (wave32 layouts, CDNA5 ISA 7.12.2)
// ---------------------------------------------------------------------------

// A fragment: 16x32 f16 tile from row-major source (ld f16 elems/row).
__device__ __forceinline__ v16h load_frag_a(const _Float16* base, int m0, int ld, int kb) {
    const int lane = threadIdx.x & 31;
    const _Float16* p = base + (m0 + (lane & 15)) * ld + kb + ((lane & 16) ? 8 : 0);
    v8h lo = *(const v8h*)(p);
    v8h hi = *(const v8h*)(p + 16);
    v16h out;
#pragma unroll
    for (int i = 0; i < 8; ++i) { out[i] = lo[i]; out[i + 8] = hi[i]; }
    return out;
}

// B fragment: 32x16 f16 tile; source stored N-major (B^T row-major: bt[n][k]).
__device__ __forceinline__ v16h load_frag_b(const _Float16* bt, int n0, int ld, int kb) {
    const int lane = threadIdx.x & 31;
    const _Float16* p = bt + (n0 + (lane & 15)) * ld + kb + ((lane & 16) ? 8 : 0);
    v8h lo = *(const v8h*)(p);
    v8h hi = *(const v8h*)(p + 16);
    v16h out;
#pragma unroll
    for (int i = 0; i < 8; ++i) { out[i] = lo[i]; out[i + 8] = hi[i]; }
    return out;
}

// D store: element (M = m0 + v + (L<16?0:8), N = n0 + (L&15)) at VGPR v.
__device__ __forceinline__ void store_d_silu_f16(_Float16* dst, int m0, int n0, int ld, v8f d) {
    const int lane = threadIdx.x & 31;
    const int nl = n0 + (lane & 15);
    const int mb = m0 + ((lane & 16) ? 8 : 0);
#pragma unroll
    for (int v = 0; v < 8; ++v) dst[(mb + v) * ld + nl] = (_Float16)silu_f(d[v]);
}

// Transposed D store: element (M,N) -> dst[N*ldT + M]  (f16, silu applied)
__device__ __forceinline__ void store_dT_silu_f16(_Float16* dst, int m0, int n0, int ldT, v8f d) {
    const int lane = threadIdx.x & 31;
    const int nl = n0 + (lane & 15);
    const int mb = m0 + ((lane & 16) ? 8 : 0);
#pragma unroll
    for (int v = 0; v < 8; ++v) dst[nl * ldT + (mb + v)] = (_Float16)silu_f(d[v]);
}

// Raw f32 D store (pre-softmax scores)
__device__ __forceinline__ void store_d_f32(float* dst, int m0, int n0, int ld, v8f d) {
    const int lane = threadIdx.x & 31;
    const int nl = n0 + (lane & 15);
    const int mb = m0 + ((lane & 16) ? 8 : 0);
#pragma unroll
    for (int v = 0; v < 8; ++v) dst[(mb + v) * ld + nl] = d[v];
}

// ---------------------------------------------------------------------------
// Prologue: convert/pad all weights to f16 in workspace.
// ---------------------------------------------------------------------------
__global__ void cvt_w_kernel(const float* __restrict__ Aq, const float* __restrict__ Ak,
                             const float* __restrict__ Av, const float* __restrict__ W_in,
                             const float* __restrict__ W_out, _Float16* __restrict__ w16) {
    int i = blockIdx.x * blockDim.x + threadIdx.x;
    if (i < kH * kH) {
        w16[kOffAq + i] = (_Float16)Aq[i];
        w16[kOffAk + i] = (_Float16)Ak[i];
        w16[kOffAv + i] = (_Float16)Av[i];
    }
    if (i < kH * kDinPad) {   // W_in: 64 rows x 32 cols, cols >= 8 zero-padded
        int r = i / kDinPad, c = i % kDinPad;
        w16[kOffWin + i] = (_Float16)(c < kDIN ? W_in[r * kDIN + c] : 0.0f);
    }
    if (i < kOutPad * kH) {   // W_out: 16 rows x 64 cols, rows >= 8 zero-padded
        int r = i / kH, c = i % kH;
        w16[kOffWout + i] = (_Float16)(r < kDOUT ? W_out[r * kH + c] : 0.0f);
    }
}

// ---------------------------------------------------------------------------
// Main: one wave32 per batch element, every matmul on the WMMA pipe.
// ---------------------------------------------------------------------------
__global__ __launch_bounds__(32) void attn_batch_kernel(
    const float* __restrict__ x, const float* __restrict__ lap,
    const float* __restrict__ b_in, const _Float16* __restrict__ w16,
    const float* __restrict__ b_out, float* __restrict__ out)
{
    __shared__ _Float16 sh_h[kNA * kH];     // h[n][j] f16; reused: attn16[n][i] (phase 5+)
    __shared__ _Float16 sh_qk[kH * kH];     // Q[i][n]|KT[i][n] (ld 32); reused: Amat[i][j] (ld 64)
    __shared__ _Float16 sh_vt[kNA * kH];    // x16 staging (phase 1) then VT[n][j] (ld 64)
    __shared__ float    sh_s[kH * kH];      // S[i][j] f32; reused: y staging (32x8 f32)
    __shared__ float    sh_lap[kNA];        // per-node laplacian
    __shared__ float    sh_inv[kH];         // per-row softmax 1/sum

    const int b    = blockIdx.x;
    const int lane = threadIdx.x;
    const _Float16* Aq16   = w16 + kOffAq;
    const _Float16* Ak16   = w16 + kOffAk;
    const _Float16* Av16   = w16 + kOffAv;
    const _Float16* Win16  = w16 + kOffWin;
    const _Float16* Wout16 = w16 + kOffWout;

    // ---- Phase 1a: stage lap and zero-padded x16 (lane == node n) ----
    const float lap_n = lap[b * kNA + lane];
    sh_lap[lane] = lap_n;
    _Float16* x16 = sh_vt;  // 32 x 32 f16, ld kDinPad
    {
        const float* xr = x + (size_t)(b * kNA + lane) * kDIN;
#pragma unroll
        for (int d = 0; d < kDIN; d += 2) {
            v2h pk; pk[0] = (_Float16)xr[d]; pk[1] = (_Float16)xr[d + 1];
            *(v2h*)&x16[lane * kDinPad + d] = pk;
        }
        v2h z; z[0] = (_Float16)0.0f; z[1] = (_Float16)0.0f;
#pragma unroll
        for (int c = kDIN; c < kDinPad; c += 2)
            *(v2h*)&x16[lane * kDinPad + c] = z;
    }
    __syncthreads();

    // ---- Phase 1b: h = silu(x @ W_in^T + b_in) * lap   M=n(32) N=j(64) K=32(pad)
#pragma unroll
    for (int mt = 0; mt < 2; ++mt) {
        v16h af = load_frag_a(x16, 16 * mt, kDinPad, 0);
#pragma unroll
        for (int nt = 0; nt < 4; ++nt) {
            v16h bf = load_frag_b(Win16, 16 * nt, kDinPad, 0);  // W_in row-major == B^T
            v8f acc = {};
            acc = __builtin_amdgcn_wmma_f32_16x16x32_f16(false, af, false, bf,
                                                         (short)0, acc, false, false);
            const int nl = 16 * nt + (lane & 15);
            const int mb = 16 * mt + ((lane & 16) ? 8 : 0);
            const float bj = b_in[nl];
#pragma unroll
            for (int v = 0; v < 8; ++v)
                sh_h[(mb + v) * kH + nl] = (_Float16)(silu_f(acc[v] + bj) * sh_lap[mb + v]);
        }
    }
    __syncthreads();

    _Float16* Q  = sh_qk;                // 64x32, ld 32
    _Float16* KT = sh_qk + kH * kNA;     // KT[i][n], ld 32

    // ---- Phase 2a: Q = silu(Aq @ h^T), V = silu(Av @ h^T) (shared h B-frags)
#pragma unroll
    for (int mt = 0; mt < 4; ++mt)
#pragma unroll
        for (int nt = 0; nt < 2; ++nt) {
            v8f aq = {}, av = {};
#pragma unroll
            for (int kt = 0; kt < 2; ++kt) {
                v16h bf  = load_frag_b(sh_h, 16 * nt, kH, 32 * kt);
                v16h aqf = load_frag_a(Aq16, 16 * mt, kH, 32 * kt);
                aq = __builtin_amdgcn_wmma_f32_16x16x32_f16(false, aqf, false, bf,
                                                            (short)0, aq, false, false);
                v16h avf = load_frag_a(Av16, 16 * mt, kH, 32 * kt);
                av = __builtin_amdgcn_wmma_f32_16x16x32_f16(false, avf, false, bf,
                                                            (short)0, av, false, false);
            }
            store_d_silu_f16(Q, 16 * mt, 16 * nt, kNA, aq);
            store_dT_silu_f16(sh_vt, 16 * mt, 16 * nt, kH, av);  // VT[n][j]
        }

    // ---- Phase 2b: K = silu(h @ Ak^T)   M=n(32) N=i(64) K=j(64) -> KT[i][n]
#pragma unroll
    for (int mt = 0; mt < 2; ++mt) {
        v16h af0 = load_frag_a(sh_h, 16 * mt, kH, 0);
        v16h af1 = load_frag_a(sh_h, 16 * mt, kH, 32);
#pragma unroll
        for (int nt = 0; nt < 4; ++nt) {
            v8f acc = {};
            acc = __builtin_amdgcn_wmma_f32_16x16x32_f16(false, af0, false,
                      load_frag_b(Ak16, 16 * nt, kH, 0),  (short)0, acc, false, false);
            acc = __builtin_amdgcn_wmma_f32_16x16x32_f16(false, af1, false,
                      load_frag_b(Ak16, 16 * nt, kH, 32), (short)0, acc, false, false);
            store_dT_silu_f16(KT, 16 * mt, 16 * nt, kNA, acc);
        }
    }
    __syncthreads();

    // ---- Phase 3: S = Q @ K             M=i(64) N=j(64) K=n(32), f32 out
#pragma unroll
    for (int mt = 0; mt < 4; ++mt) {
        v16h af = load_frag_a(Q, 16 * mt, kNA, 0);
#pragma unroll
        for (int nt = 0; nt < 4; ++nt) {
            v8f acc = {};
            acc = __builtin_amdgcn_wmma_f32_16x16x32_f16(false, af, false,
                      load_frag_b(KT, 16 * nt, kNA, 0), (short)0, acc, false, false);
            store_d_f32(sh_s, 16 * mt, 16 * nt, kH, acc);
        }
    }
    __syncthreads();

    // ---- Phase 4: single-pass softmax -> unnormalized exp (f16) + 1/sum per row
    _Float16* Amat = sh_qk;  // 64x64, ld 64 (overwrites Q/KT)
    for (int r = lane; r < kH; r += kNA) {
        const float* srow = sh_s + r * kH;
        float mx = srow[0];
        for (int j = 1; j < kH; ++j) mx = fmaxf(mx, srow[j]);
        float ssum = 0.0f;
        for (int j = 0; j < kH; ++j) {
            const float e = __expf(srow[j] - mx);
            Amat[r * kH + j] = (_Float16)e;
            ssum += e;
        }
        sh_inv[r] = __builtin_amdgcn_rcpf(ssum);
    }
    __syncthreads();

    // ---- Phase 5: attn = silu((E @ V) * inv_row)  M=i(64) N=n(32) K=j(64)
    _Float16* attn16 = sh_h;  // reuse: attn16[n][i], ld 64 (A-source for phase 6)
#pragma unroll
    for (int mt = 0; mt < 4; ++mt) {
        v16h af0 = load_frag_a(Amat, 16 * mt, kH, 0);
        v16h af1 = load_frag_a(Amat, 16 * mt, kH, 32);
        const int mb = 16 * mt + ((lane & 16) ? 8 : 0);
#pragma unroll
        for (int nt = 0; nt < 2; ++nt) {
            v8f acc = {};
            acc = __builtin_amdgcn_wmma_f32_16x16x32_f16(false, af0, false,
                      load_frag_b(sh_vt, 16 * nt, kH, 0),  (short)0, acc, false, false);
            acc = __builtin_amdgcn_wmma_f32_16x16x32_f16(false, af1, false,
                      load_frag_b(sh_vt, 16 * nt, kH, 32), (short)0, acc, false, false);
            const int nl = 16 * nt + (lane & 15);
#pragma unroll
            for (int v = 0; v < 8; ++v)
                attn16[nl * kH + (mb + v)] =
                    (_Float16)silu_f(acc[v] * sh_inv[mb + v]);
        }
    }
    __syncthreads();

    // ---- Phase 6: y = silu(attn @ W_out^T + b_out) * lap   M=n(32) N=o(16 pad) K=i(64)
    float* yst = sh_s;  // 32 x kDOUT f32 staging
#pragma unroll
    for (int mt = 0; mt < 2; ++mt) {
        v16h af0 = load_frag_a(attn16, 16 * mt, kH, 0);
        v16h af1 = load_frag_a(attn16, 16 * mt, kH, 32);
        v8f acc = {};
        acc = __builtin_amdgcn_wmma_f32_16x16x32_f16(false, af0, false,
                  load_frag_b(Wout16, 0, kH, 0),  (short)0, acc, false, false);
        acc = __builtin_amdgcn_wmma_f32_16x16x32_f16(false, af1, false,
                  load_frag_b(Wout16, 0, kH, 32), (short)0, acc, false, false);
        const int nl = lane & 15;
        const int mb = 16 * mt + ((lane & 16) ? 8 : 0);
        if (nl < kDOUT) {
            const float bo = b_out[nl];
#pragma unroll
            for (int v = 0; v < 8; ++v)
                yst[(mb + v) * kDOUT + nl] = silu_f(acc[v] + bo) * sh_lap[mb + v];
        }
    }
    __syncthreads();

    // ---- Phase 7: mean over nodes ----
    if (lane < kDOUT) {
        float s = 0.0f;
        for (int n = 0; n < kNA; ++n) s += yst[n * kDOUT + lane];
        out[(size_t)b * kDOUT + lane] = s * (1.0f / kNA);
    }
}

extern "C" void kernel_launch(void* const* d_in, const int* in_sizes, int n_in,
                              void* d_out, int out_size, void* d_ws, size_t ws_size,
                              hipStream_t stream) {
    const float* x     = (const float*)d_in[0];
    const float* lapl  = (const float*)d_in[1];
    const float* W_in  = (const float*)d_in[2];
    const float* b_in  = (const float*)d_in[3];
    const float* Aq    = (const float*)d_in[4];
    const float* Ak    = (const float*)d_in[5];
    const float* Av    = (const float*)d_in[6];
    const float* W_out = (const float*)d_in[7];
    const float* b_out = (const float*)d_in[8];
    float* out = (float*)d_out;

    _Float16* w16 = (_Float16*)d_ws;   // padded f16 weights, ~30 KB
    const int B = in_sizes[0] / (kNA * kDIN);  // 8192

    cvt_w_kernel<<<(kH * kH + 255) / 256, 256, 0, stream>>>(Aq, Ak, Av, W_in, W_out, w16);
    attn_batch_kernel<<<B, 32, 0, stream>>>(x, lapl, b_in, w16, b_out, out);
}
